// Encoder_38500086842058
// MI455X (gfx1250) — matrix operand order repaired and checked by reading, not compiled, not run on hardware
//
#include <hip/hip_runtime.h>
#include <stdint.h>

// ---------------- problem dims ----------------
#define Bm   32
#define Tm   512
#define Im   512
#define Hm   512
#define G3H  1536            // 3*H
#define Rm   (Bm*Tm)         // 16384 rows for the input GEMMs
#define HP   520             // padded LDS row length (bf16 elems); 520*2 % 16 == 0

typedef __bf16 bf16_t;
typedef bf16_t v16bf __attribute__((ext_vector_type(16)));
typedef float  v8f   __attribute__((ext_vector_type(8)));

union BF { v16bf v; uint4 q[2]; };

__device__ __forceinline__ v8f v8f_zero() {
    v8f z = {0.f,0.f,0.f,0.f,0.f,0.f,0.f,0.f};
    return z;
}

__device__ __forceinline__ unsigned short f2bf(float f) {
    union { float f; uint32_t u; } x; x.f = f;
    uint32_t u = x.u;
    uint32_t r = u + 0x7FFFu + ((u >> 16) & 1u);   // round-to-nearest-even
    return (unsigned short)(r >> 16);
}

// fast sigmoid/tanh: v_exp_f32 + v_rcp_f32, no div_scale/Newton chain
__device__ __forceinline__ float sigmoidf_(float x) {
    return __builtin_amdgcn_rcpf(1.0f + __expf(-x));
}
__device__ __forceinline__ float tanhf_(float x) {
    return 2.0f * __builtin_amdgcn_rcpf(1.0f + __expf(-2.0f * x)) - 1.0f;
}

// ---------------- f32 -> bf16 cast (vectorized) ----------------
__global__ void k_cast_bf16(const float* __restrict__ src,
                            unsigned short* __restrict__ dst, int n) {
    int i = (blockIdx.x * blockDim.x + threadIdx.x) * 4;
    if (i + 3 < n) {
        float4 f = *(const float4*)(src + i);
        ushort4 o;
        o.x = f2bf(f.x); o.y = f2bf(f.y); o.z = f2bf(f.z); o.w = f2bf(f.w);
        *(ushort4*)(dst + i) = o;
    }
}

// ---------------- pack weight (3H x K, row-major f32) into per-lane WMMA
// B-fragment layout: dst[nt][kc][lane][16 halfs], B(k,n) = W[n][k].
// 16-bit B 32x16: lanes 0-15 hold K=0..7,16..23; lanes 16-31 hold K=8..15,24..31.
__global__ void k_pack_w(const float* __restrict__ W,
                         unsigned short* __restrict__ dst, int K, int KC) {
    int idx = blockIdx.x * blockDim.x + threadIdx.x;
    int total = 96 * KC * 32;                      // 96 column tiles (3H/16)
    if (idx >= total) return;
    int lane = idx & 31;
    int rest = idx >> 5;
    int kc   = rest % KC;
    int nt   = rest / KC;
    int n     = nt * 16 + (lane & 15);
    int kbase = kc * 32 + (lane >> 4) * 8;
    unsigned short* out = dst + (size_t)((nt * KC + kc) * 32 + lane) * 16;
#pragma unroll
    for (int e = 0; e < 16; ++e)
        out[e] = f2bf(W[(size_t)n * K + (kbase + (e < 8 ? e : e + 8))]);
}

// ---------------- input GEMM: C(R x 3H) = A(R x K, bf16) * Bp + bias ------
// 8 waves/block; each wave computes one 16-row tile x eight 16-col tiles.
// KCt is compile-time so all fragment offsets are VMEM immediates off one
// base pointer per operand (+64B/kc for A, +1024B/kc for B).
template <int KCt>
__global__ __launch_bounds__(256) void k_gemm_pre(
    const unsigned short* __restrict__ A,
    const unsigned short* __restrict__ Bp,
    const float* __restrict__ bias,
    float* __restrict__ C) {
    constexpr int K = KCt * 32;
    const int lane = threadIdx.x & 31;
    const int wave = threadIdx.x >> 5;
    const int gw   = blockIdx.x * 8 + wave;
    const int rt   = gw / 12;                 // row tile 0..1023
    const int ct0  = (gw % 12) * 8;           // first of 8 col tiles
    const int lmod = lane & 15;
    const int hi8  = (lane >> 4) * 8;
    const int arow = rt * 16 + lmod;

    const char* ap = (const char*)A + ((size_t)arow * K + hi8) * 2;        // +64/kc
    const char* bp = (const char*)Bp + (size_t)ct0 * (KCt * 1024)
                                     + (size_t)lane * 32;                  // +1024/kc

    v8f acc[8];
#pragma unroll
    for (int j = 0; j < 8; ++j) acc[j] = v8f_zero();

    for (int kc = 0; kc < KCt; ++kc) {
        BF a;
        a.q[0] = *(const uint4*)ap;            // K = kc*32 + hi8 .. +7
        a.q[1] = *(const uint4*)(ap + 32);     // K = kc*32 + 16 + hi8 .. +7
        BF b[2];
        b[0].q[0] = *(const uint4*)bp;
        b[0].q[1] = *(const uint4*)(bp + 16);
#pragma unroll
        for (int j = 0; j < 8; ++j) {
            if (j < 7) {                       // prefetch next tile (const offs)
                const char* nb = bp + (size_t)((j + 1) * (KCt * 1024));
                b[(j + 1) & 1].q[0] = *(const uint4*)nb;
                b[(j + 1) & 1].q[1] = *(const uint4*)(nb + 16);
            }
            acc[j] = __builtin_amdgcn_wmma_f32_16x16x32_bf16(
                false, a.v, false, b[j & 1].v, (short)0, acc[j], false, false);
        }
        ap += 64;
        bp += 1024;
    }
#pragma unroll
    for (int j = 0; j < 8; ++j) {
        const int col = (ct0 + j) * 16 + lmod;
        const float bv = bias[col];
#pragma unroll
        for (int vv = 0; vv < 8; ++vv) {           // C layout: M = v + hi8, N = lmod
            int r = rt * 16 + vv + hi8;
            C[(size_t)r * G3H + col] = acc[j][vv] + bv;
        }
    }
}

// ---------------- persistent GRU scan: one block per direction ----------
// 32 waves; wave owns (mt = wave/16, 2 column groups ct0, ct0+1); for each
// group it computes the r/z/n tiles (same hidden columns), so the gate math
// and h update are wave-local. h lives in registers (f32) + LDS (bf16).
// W_hh is re-streamed from L2 every step; the opaque 32-bit offset prevents
// LICM from hoisting 3KB/lane of weights out of the time loop (which spills)
// while preserving the global address space of the loads.
__global__ __launch_bounds__(1024) void k_gru_scan(
    const float* __restrict__ preF, const float* __restrict__ preB,
    const unsigned short* __restrict__ WhhF, const unsigned short* __restrict__ WhhB,
    const float* __restrict__ bhhF, const float* __restrict__ bhhB,
    unsigned short* __restrict__ y_bf, float* __restrict__ y_f32,
    float* __restrict__ finals, int layer, int out_bf16) {
    const int dir = blockIdx.x;
    const float* pre          = dir ? preB : preF;
    const unsigned short* Whh = dir ? WhhB : WhhF;
    const float* bhh          = dir ? bhhB : bhhF;

    __shared__ unsigned short hbf[32 * HP];        // h in bf16, 33 KB

    const int lane = threadIdx.x & 31;
    const int wave = threadIdx.x >> 5;
    const int mt   = wave >> 4;                    // row tile (batch 0-15 / 16-31)
    const int ct0  = (wave & 15) * 2;              // hidden col tiles ct0, ct0+1
    const int hi8  = (lane >> 4) * 8;
    const int lmod = lane & 15;
    const int arow = mt * 16 + lmod;

    // byte offset of tile u (= g*3+gate) relative to wp:  g*16KB + gate*512KB, plus 128 B for u=5
    constexpr int OFF[6] = {0, 524288, 1048576, 16384, 540672, 1065088};

    const char* hp0 = (const char*)&hbf[arow * HP + hi8];            // +64/kc
    const char* wp0 = (const char*)Whh + (size_t)(ct0 * 16) * 1024
                                       + (size_t)lane * 32;          // +1024/kc

    for (int i = threadIdx.x; i < 32 * HP; i += blockDim.x) hbf[i] = 0;
    v8f h[2]; h[0] = v8f_zero(); h[1] = v8f_zero();
    __syncthreads();

#pragma unroll 1
    for (int s = 0; s < Tm; ++s) {
        const int t = dir ? (Tm - 1 - s) : s;      // bwd == reversed-time scan
        v8f acc[6];
#pragma unroll
        for (int i = 0; i < 6; ++i) acc[i] = v8f_zero();

        // Opaque zero offset: defeats LICM for the (step-invariant) weight
        // loads without erasing pointer provenance -> stays global_load_b128.
        unsigned wo = 0;
        asm volatile("" : "+v"(wo));
        const char* hp = hp0;
        const char* wp = wp0 + wo;
#pragma unroll 1
        for (int kc = 0; kc < 16; ++kc) {          // K = 512
            BF a;                                  // two ds_load_b128
            a.q[0] = *(const uint4*)hp;
            a.q[1] = *(const uint4*)(hp + 32);
            BF b[2];
            b[0].q[0] = *(const uint4*)(wp + OFF[0]);
            b[0].q[1] = *(const uint4*)(wp + OFF[0] + 16);
#pragma unroll
            for (int u = 0; u < 6; ++u) {          // u = g*3 + gate
                if (u < 5) {
                    b[(u + 1) & 1].q[0] = *(const uint4*)(wp + OFF[u + 1]);
                    b[(u + 1) & 1].q[1] = *(const uint4*)(wp + OFF[u + 1] + 16);
                }
                acc[u] = __builtin_amdgcn_wmma_f32_16x16x32_bf16(
                    false, a.v, false, b[u & 1].v, (short)0, acc[u], false, false);
            }
            hp += 64;
            wp += 1024;
        }
        __syncthreads();                           // all reads of hbf done

#pragma unroll
        for (int g = 0; g < 2; ++g) {
            const int ct = ct0 + g;
            const int colbase = ct * 16 + lmod;
            const float br = bhh[colbase];
            const float bz = bhh[512 + colbase];
            const float bn = bhh[1024 + colbase];
#pragma unroll
            for (int vv = 0; vv < 8; ++vv) {
                const int b = mt * 16 + vv + hi8;
                const size_t gxb = ((size_t)b * Tm + t) * G3H;
                const float gxr = pre[gxb + colbase];
                const float gxz = pre[gxb + 512 + colbase];
                const float gxn = pre[gxb + 1024 + colbase];
                const float ghr = acc[g * 3 + 0][vv] + br;
                const float ghz = acc[g * 3 + 1][vv] + bz;
                const float ghn = acc[g * 3 + 2][vv] + bn;
                const float r  = sigmoidf_(gxr + ghr);
                const float z  = sigmoidf_(gxz + ghz);
                const float nn = tanhf_(gxn + r * ghn);
                const float hn = (1.0f - z) * nn + z * h[g][vv];
                h[g][vv] = hn;
                hbf[b * HP + colbase] = f2bf(hn);
                const size_t yi = ((size_t)b * Tm + t) * 1024 + (size_t)dir * 512 + colbase;
                if (out_bf16) y_bf[yi] = f2bf(hn);
                else          y_f32[yi] = hn;
            }
        }
        __syncthreads();                           // hbf writes visible
    }

    // final hidden state -> d_out final section: [layer][b][dir*H + c]
#pragma unroll
    for (int g = 0; g < 2; ++g) {
        const int colbase = (ct0 + g) * 16 + lmod;
#pragma unroll
        for (int vv = 0; vv < 8; ++vv) {
            const int b = mt * 16 + vv + hi8;
            finals[(size_t)layer * Bm * 1024 + (size_t)b * 1024 +
                   (size_t)dir * 512 + colbase] = h[g][vv];
        }
    }
}

extern "C" void kernel_launch(void* const* d_in, const int* in_sizes, int n_in,
                              void* d_out, int out_size, void* d_ws, size_t ws_size,
                              hipStream_t stream) {
    (void)in_sizes; (void)n_in; (void)out_size; (void)ws_size;
    const float* ctx      = (const float*)d_in[0];
    const float* w_ih_l0  = (const float*)d_in[1];
    const float* w_hh_l0  = (const float*)d_in[2];
    const float* b_ih_l0  = (const float*)d_in[3];
    const float* b_hh_l0  = (const float*)d_in[4];
    const float* w_ih_l0r = (const float*)d_in[5];
    const float* w_hh_l0r = (const float*)d_in[6];
    const float* b_ih_l0r = (const float*)d_in[7];
    const float* b_hh_l0r = (const float*)d_in[8];
    const float* w_ih_l1  = (const float*)d_in[9];
    const float* w_hh_l1  = (const float*)d_in[10];
    const float* b_ih_l1  = (const float*)d_in[11];
    const float* b_hh_l1  = (const float*)d_in[12];
    const float* w_ih_l1r = (const float*)d_in[13];
    const float* w_hh_l1r = (const float*)d_in[14];
    const float* b_ih_l1r = (const float*)d_in[15];
    const float* b_hh_l1r = (const float*)d_in[16];

    char* ws = (char*)d_ws;
    size_t off = 0;
    auto alloc = [&](size_t bytes) -> void* {
        void* p = ws + off;
        off += (bytes + 255) & ~(size_t)255;
        return p;
    };
    unsigned short* ctx_bf = (unsigned short*)alloc((size_t)Rm * Im * 2);      // 16.8 MB
    unsigned short* x1_bf  = (unsigned short*)alloc((size_t)Rm * 1024 * 2);    // 33.6 MB
    float* preF            = (float*)alloc((size_t)Rm * G3H * 4);              // 100.7 MB (reused l0/l1)
    float* preB            = (float*)alloc((size_t)Rm * G3H * 4);              // 100.7 MB
    unsigned short* p_wih0  = (unsigned short*)alloc((size_t)96 * 16 * 32 * 16 * 2);
    unsigned short* p_wih0r = (unsigned short*)alloc((size_t)96 * 16 * 32 * 16 * 2);
    unsigned short* p_wih1  = (unsigned short*)alloc((size_t)96 * 32 * 32 * 16 * 2);
    unsigned short* p_wih1r = (unsigned short*)alloc((size_t)96 * 32 * 32 * 16 * 2);
    unsigned short* p_whh0  = (unsigned short*)alloc((size_t)96 * 16 * 32 * 16 * 2);
    unsigned short* p_whh0r = (unsigned short*)alloc((size_t)96 * 16 * 32 * 16 * 2);
    unsigned short* p_whh1  = (unsigned short*)alloc((size_t)96 * 16 * 32 * 16 * 2);
    unsigned short* p_whh1r = (unsigned short*)alloc((size_t)96 * 16 * 32 * 16 * 2);

    float* outx   = (float*)d_out;
    float* finals = outx + (size_t)Rm * 1024;      // after x (B,T,2H)

    // 1) cast context to bf16
    k_cast_bf16<<<(Rm * Im) / 1024, 256, 0, stream>>>(ctx, ctx_bf, Rm * Im);

    // 2) pack all weights into WMMA B-fragment layout
    k_pack_w<<<192, 256, 0, stream>>>(w_ih_l0,  p_wih0,  512, 16);
    k_pack_w<<<192, 256, 0, stream>>>(w_ih_l0r, p_wih0r, 512, 16);
    k_pack_w<<<384, 256, 0, stream>>>(w_ih_l1,  p_wih1,  1024, 32);
    k_pack_w<<<384, 256, 0, stream>>>(w_ih_l1r, p_wih1r, 1024, 32);
    k_pack_w<<<192, 256, 0, stream>>>(w_hh_l0,  p_whh0,  512, 16);
    k_pack_w<<<192, 256, 0, stream>>>(w_hh_l0r, p_whh0r, 512, 16);
    k_pack_w<<<192, 256, 0, stream>>>(w_hh_l1,  p_whh1,  512, 16);
    k_pack_w<<<192, 256, 0, stream>>>(w_hh_l1r, p_whh1r, 512, 16);

    // 3) layer-0 input projections (both directions). 12288 waves = 1024 row
    //    tiles x 12 col groups; 8 waves/block -> 1536 blocks.
    k_gemm_pre<16><<<1536, 256, 0, stream>>>(ctx_bf, p_wih0,  b_ih_l0,  preF);
    k_gemm_pre<16><<<1536, 256, 0, stream>>>(ctx_bf, p_wih0r, b_ih_l0r, preB);

    // 4) layer-0 scan (writes x1 as bf16, finals for layer 0)
    k_gru_scan<<<2, 1024, 0, stream>>>(preF, preB, p_whh0, p_whh0r, b_hh_l0, b_hh_l0r,
                                       x1_bf, nullptr, finals, 0, 1);

    // 5) layer-1 input projections (K = 2H = 1024), reuse pre buffers
    k_gemm_pre<32><<<1536, 256, 0, stream>>>(x1_bf, p_wih1,  b_ih_l1,  preF);
    k_gemm_pre<32><<<1536, 256, 0, stream>>>(x1_bf, p_wih1r, b_ih_l1r, preB);

    // 6) layer-1 scan (writes y directly into d_out as f32, finals for layer 1)
    k_gru_scan<<<2, 1024, 0, stream>>>(preF, preB, p_whh1, p_whh1r, b_hh_l1, b_hh_l1r,
                                       nullptr, outx, finals, 1, 0);
}
